// SequentialGraphConvolutionalNetwork_7035156431280
// MI455X (gfx1250) — matrix operand order repaired and checked by reading, not compiled, not run on hardware
//
#include <hip/hip_runtime.h>
#include <hip/hip_bf16.h>

// ---------------------------------------------------------------------------
// Types for CDNA5 WMMA (wave32): v_wmma_f32_16x16x32_bf16
// ---------------------------------------------------------------------------
typedef __attribute__((ext_vector_type(16))) __bf16 v16bf;
typedef __attribute__((ext_vector_type(8)))  __bf16 v8bf;
typedef __attribute__((ext_vector_type(8)))  float  v8f;

#define NNODES 4096
#define DIN    1024
#define GC     1024   // GC1 == GC2 == 1024
#define FC1N   512
#define NCLS   2

__device__ __forceinline__ __bf16 f2bf(float f) {
    unsigned int u = __builtin_bit_cast(unsigned int, f);
    unsigned int r = u + 0x7FFFu + ((u >> 16) & 1u);   // round-to-nearest-even
    unsigned short h = (unsigned short)(r >> 16);
    return __builtin_bit_cast(__bf16, h);
}

__device__ __forceinline__ float selu_f(float x) {
    const float scale = 1.0507009873554805f;
    const float alpha = 1.6732632423543772f;
    return x > 0.0f ? scale * x : scale * alpha * (expf(x) - 1.0f);
}

// ---------------------------------------------------------------------------
// Elementwise fp32 -> bf16
// ---------------------------------------------------------------------------
__global__ void k_f32_to_bf16(const float* __restrict__ in, __bf16* __restrict__ out, int n) {
    int i = blockIdx.x * blockDim.x + threadIdx.x;
    if (i < n) out[i] = f2bf(in[i]);
}

// W: [K][Nc] fp32 row-major  ->  Wt: [Nc][K] bf16 row-major (N-major for B frags)
__global__ void k_transpose_bf16(const float* __restrict__ W, __bf16* __restrict__ Wt,
                                 int K, int Nc) {
    int idx = blockIdx.x * blockDim.x + threadIdx.x;
    if (idx >= K * Nc) return;
    int n = idx / K;
    int k = idx - n * K;
    Wt[idx] = f2bf(W[k * Nc + n]);
}

// ---------------------------------------------------------------------------
// dinv[i] = (row degree of banded A)^{-1/2}
// ---------------------------------------------------------------------------
__global__ void k_degree(const float* __restrict__ fw, const float* __restrict__ bw,
                         float* __restrict__ dinv, int n) {
    int i = blockIdx.x * blockDim.x + threadIdx.x;
    if (i >= n) return;
    float d = 1.0f;
    #pragma unroll
    for (int j = 1; j <= 8; ++j) {
        if (i + j < n)  d += fw[j - 1];
        if (i - j >= 0) d += bw[j - 1];
    }
    dinv[i] = rsqrtf(d);
}

// ---------------------------------------------------------------------------
// WMMA bf16 GEMM:  C[M,N] = A[M,K] (bf16, row-major) * Bt[N,K]^T (bf16) + bias[N]
// Block = 128 threads (4 waves). Each wave computes a 64x64 tile (4x4 frags).
// Grid = (N/128, M/128).
// ---------------------------------------------------------------------------
__device__ __forceinline__ v16bf load_a_frag(const __bf16* __restrict__ A, int lda,
                                             int m0, int kb, int lane) {
    // A frag, contiguous k order (same as B): lanes 0-15 row M=lane, K = kb..kb+15
    //                        lanes16-31 row M=lane-16, K = kb+16..kb+31
    int row  = m0 + (lane & 15);
    int kofs = kb + ((lane & 16) ? 16 : 0);
    const __bf16* p = A + (size_t)row * lda + kofs;
    v16bf r = *(const v16bf*)(p);
    return r;
}

__device__ __forceinline__ v16bf load_b_frag(const __bf16* __restrict__ Bt, int ldb,
                                             int n0, int kb, int lane) {
    // B frag, contiguous k order (same as A): col N = lane%16; lanes 0-15 K=kb..kb+15, lanes 16-31 K=kb+16..kb+31
    int col  = n0 + (lane & 15);
    int kofs = kb + ((lane & 16) ? 16 : 0);
    return *(const v16bf*)(Bt + (size_t)col * ldb + kofs);
}

__global__ void __launch_bounds__(128)
k_gemm_bf16(const __bf16* __restrict__ A, const __bf16* __restrict__ Bt,
            const float* __restrict__ bias, float* __restrict__ C,
            int M, int N, int K) {
    const int lane = threadIdx.x & 31;
    const int wid  = threadIdx.x >> 5;            // 0..3
    const int m0   = blockIdx.y * 128 + (wid >> 1) * 64;
    const int n0   = blockIdx.x * 128 + (wid & 1) * 64;

    v8f acc[4][4] = {};

    for (int kb = 0; kb < K; kb += 32) {
        v16bf af[4], bf[4];
        #pragma unroll
        for (int i = 0; i < 4; ++i) af[i] = load_a_frag(A, K, m0 + 16 * i, kb, lane);
        #pragma unroll
        for (int j = 0; j < 4; ++j) bf[j] = load_b_frag(Bt, K, n0 + 16 * j, kb, lane);
        #pragma unroll
        for (int i = 0; i < 4; ++i)
            #pragma unroll
            for (int j = 0; j < 4; ++j)
                acc[i][j] = __builtin_amdgcn_wmma_f32_16x16x32_bf16(
                    false, af[i], false, bf[j], (short)0, acc[i][j], false, false);
    }

    // C/D layout: col = lane%16, row = m0 + 16*i + r + (lane>=16 ? 8 : 0)
    const int colin = lane & 15;
    const int rbase = (lane & 16) ? 8 : 0;
    #pragma unroll
    for (int j = 0; j < 4; ++j) {
        int col = n0 + 16 * j + colin;
        float b = bias[col];
        #pragma unroll
        for (int i = 0; i < 4; ++i) {
            #pragma unroll
            for (int r = 0; r < 8; ++r)
                C[(size_t)(m0 + 16 * i + rbase + r) * N + col] = acc[i][j][r] + b;
        }
    }
}

// ---------------------------------------------------------------------------
// Banded A multiply + SELU.  out[row,:] = selu( sum_{k=-8..8} coef_k * P[row+k,:] )
// One block per row; 17 normalized coefficients staged in LDS.
// ---------------------------------------------------------------------------
template <bool OUT_BF16>
__global__ void __launch_bounds__(256)
k_band_selu(const float* __restrict__ P, const float* __restrict__ dinv,
            const float* __restrict__ fw, const float* __restrict__ bw,
            void* __restrict__ out, int n, int cols) {
    const int row = blockIdx.x;
    __shared__ float coef[17];
    if (threadIdx.x < 17) {
        int k = (int)threadIdx.x - 8;
        int j = row + k;
        float c = 0.0f;
        if (j >= 0 && j < n) {
            float a = (k == 0) ? 1.0f : (k > 0 ? fw[k - 1] : bw[-k - 1]);
            c = dinv[row] * a * dinv[j];
        }
        coef[threadIdx.x] = c;
    }
    __syncthreads();

    for (int col = threadIdx.x; col < cols; col += blockDim.x) {
        float accv = 0.0f;
        #pragma unroll
        for (int k = -8; k <= 8; ++k) {
            int j = row + k;
            if (j >= 0 && j < n)
                accv += coef[k + 8] * P[(size_t)j * cols + col];
        }
        float s = selu_f(accv);
        if (OUT_BF16) ((__bf16*)out)[(size_t)row * cols + col] = f2bf(s);
        else          ((float*)out)[(size_t)row * cols + col] = s;
    }
}

// ---------------------------------------------------------------------------
// Column mean over rows: mean[c] = (1/rows) * sum_r H[r,c]
// ---------------------------------------------------------------------------
__global__ void k_colmean(const float* __restrict__ H, float* __restrict__ mean,
                          int rows, int cols) {
    int c = blockIdx.x * blockDim.x + threadIdx.x;
    if (c >= cols) return;
    float s = 0.0f;
    for (int r = 0; r < rows; ++r) s += H[(size_t)r * cols + c];
    mean[c] = s * (1.0f / (float)rows);
}

// ---------------------------------------------------------------------------
// FC head: z = selu(pooled @ W1 + b1); out = z @ W2 + b2.  One block, 512 thr.
// ---------------------------------------------------------------------------
__global__ void __launch_bounds__(512)
k_head(const float* __restrict__ pooled,
       const float* __restrict__ w1, const float* __restrict__ b1,
       const float* __restrict__ w2, const float* __restrict__ b2,
       float* __restrict__ out) {
    __shared__ float z[FC1N];
    int j = threadIdx.x;                       // 0..511
    float acc = b1[j];
    for (int k = 0; k < GC; ++k) acc += pooled[k] * w1[k * FC1N + j];
    z[j] = selu_f(acc);
    __syncthreads();
    if (j < NCLS) {
        float o = b2[j];
        for (int k = 0; k < FC1N; ++k) o += z[k] * w2[k * NCLS + j];
        out[j] = o;
    }
}

// ---------------------------------------------------------------------------
// Launch
// ---------------------------------------------------------------------------
extern "C" void kernel_launch(void* const* d_in, const int* in_sizes, int n_in,
                              void* d_out, int out_size, void* d_ws, size_t ws_size,
                              hipStream_t stream) {
    const float* x     = (const float*)d_in[0];
    const float* fw_w  = (const float*)d_in[1];
    const float* bw_w  = (const float*)d_in[2];
    const float* gc_w1 = (const float*)d_in[3];
    const float* gc_b1 = (const float*)d_in[4];
    const float* gc_w2 = (const float*)d_in[5];
    const float* gc_b2 = (const float*)d_in[6];
    const float* fc_w1 = (const float*)d_in[7];
    const float* fc_b1 = (const float*)d_in[8];
    const float* fc_w2 = (const float*)d_in[9];
    const float* fc_b2 = (const float*)d_in[10];
    float* outp = (float*)d_out;

    // Workspace layout (bytes)
    char* ws = (char*)d_ws;
    const size_t SZ_XB  = (size_t)NNODES * DIN * 2;   // 8 MB  bf16 X
    const size_t SZ_WT  = (size_t)DIN * GC * 2;       // 2 MB  bf16 W^T each
    const size_t SZ_H1B = (size_t)NNODES * GC * 2;    // 8 MB  bf16 band1 out
    const size_t SZ_P   = (size_t)NNODES * GC * 4;    // 16 MB fp32 gemm out (reused)
    const size_t SZ_H2  = (size_t)NNODES * GC * 4;    // 16 MB fp32 band2 out
    __bf16* Xb   = (__bf16*)(ws);
    __bf16* W1t  = (__bf16*)(ws + SZ_XB);
    __bf16* W2t  = (__bf16*)(ws + SZ_XB + SZ_WT);
    __bf16* H1b  = (__bf16*)(ws + SZ_XB + 2 * SZ_WT);
    float*  P    = (float*) (ws + SZ_XB + 2 * SZ_WT + SZ_H1B);
    float*  H2   = (float*) (ws + SZ_XB + 2 * SZ_WT + SZ_H1B + SZ_P);
    float*  dinv = (float*) (ws + SZ_XB + 2 * SZ_WT + SZ_H1B + SZ_P + SZ_H2);
    float*  mean = dinv + NNODES;

    // 1) conversions
    {
        int n = NNODES * DIN;
        k_f32_to_bf16<<<(n + 255) / 256, 256, 0, stream>>>(x, Xb, n);
        int m = DIN * GC;
        k_transpose_bf16<<<(m + 255) / 256, 256, 0, stream>>>(gc_w1, W1t, DIN, GC);
        k_transpose_bf16<<<(m + 255) / 256, 256, 0, stream>>>(gc_w2, W2t, GC, GC);
    }
    // 2) degree normalization
    k_degree<<<(NNODES + 255) / 256, 256, 0, stream>>>(fw_w, bw_w, dinv, NNODES);

    // 3) GEMM1: P = Xb @ W1 + b1
    {
        dim3 grid(GC / 128, NNODES / 128);
        k_gemm_bf16<<<grid, 128, 0, stream>>>(Xb, W1t, gc_b1, P, NNODES, GC, DIN);
    }
    // 4) band + selu -> bf16 H1b
    k_band_selu<true><<<NNODES, 256, 0, stream>>>(P, dinv, fw_w, bw_w, (void*)H1b, NNODES, GC);

    // 5) GEMM2: P = H1b @ W2 + b2
    {
        dim3 grid(GC / 128, NNODES / 128);
        k_gemm_bf16<<<grid, 128, 0, stream>>>(H1b, W2t, gc_b2, P, NNODES, GC, GC);
    }
    // 6) band + selu -> fp32 H2
    k_band_selu<false><<<NNODES, 256, 0, stream>>>(P, dinv, fw_w, bw_w, (void*)H2, NNODES, GC);

    // 7) mean pooling over nodes
    k_colmean<<<(GC + 255) / 256, 256, 0, stream>>>(H2, mean, NNODES, GC);

    // 8) FC head -> logits
    k_head<<<1, FC1N, 0, stream>>>(mean, fc_w1, fc_b1, fc_w2, fc_b2, outp);
}
